// CrossMdoalBlock_70755291234562
// MI455X (gfx1250) — matrix operand order repaired and checked by reading, not compiled, not run on hardware
//
#include <hip/hip_runtime.h>
#include <hip/hip_bf16.h>

// ---------------------------------------------------------------------------
// CDNA5 / gfx1250 cross-modal block.
// All matmuls on v_wmma_f32_16x16x32_bf16. GEMM staging now uses
// GLOBAL_LOAD_ASYNC_TO_LDS_B128 (ASYNCcnt) with double-buffered LDS tiles:
// no VGPR round-trip, async copies overlap the WMMA of the previous tile.
// Attention: probs = 1 - softmax  =>  ctx = colsum(V) - flashattn(Q,K,V).
// BiGRU: persistent per-direction workgroup, h + gate accums in LDS.
// ---------------------------------------------------------------------------

typedef __bf16 bf16;
typedef __attribute__((ext_vector_type(16))) __bf16 v16bf;
typedef __attribute__((ext_vector_type(8)))  float  v8f;

union F16x16 { v16bf v; uint4 q[2]; };
union CDu { v8f v; float e[8]; };

__device__ inline v8f wmma_bf16(v16bf a, v16bf b, v8f c) {
  return __builtin_amdgcn_wmma_f32_16x16x32_bf16(
      false, a, false, b, (short)0, c, false, false);
}

// Direct global -> LDS async copy, 16B per lane (ISA 15.18.3 op 98, ASYNCcnt).
// Generic pointers to LDS hold the DS offset in the low 32 bits (aperture
// layout, ISA 10.2), so truncation yields the VDST DS address.
__device__ inline void async_copy_b128(const bf16* g, bf16* l) {
  unsigned laddr = (unsigned)(uintptr_t)l;
  asm volatile("global_load_async_to_lds_b128 %0, %1, off"
               :: "v"(laddr), "v"(g)
               : "memory");
}
__device__ inline void wait_async0() {
  asm volatile("s_wait_asynccnt 0" ::: "memory");
}

// A fragment (16x32 MxK) from row-major base[m*ld + k].
__device__ inline v16bf load_frag_a(const bf16* base, int ld) {
  int lane = threadIdx.x & 31;
  int m = lane & 15, h = lane >> 4;
  const char* p = (const char*)(base + m * ld) + (h << 4);
  F16x16 r;
  r.q[0] = *(const uint4*)(p);
  r.q[1] = *(const uint4*)(p + 32);
  return r.v;
}

// B fragment (32x16 KxN) where source is [N,K]-major: value(k,n)=base[n*ld+k].
__device__ inline v16bf load_frag_bk(const bf16* base, int ld) {
  int lane = threadIdx.x & 31;
  int n = lane & 15, h = lane >> 4;
  const char* p = (const char*)(base + n * ld) + (h << 4);
  F16x16 r;
  r.q[0] = *(const uint4*)(p);
  r.q[1] = *(const uint4*)(p + 32);
  return r.v;
}

// ---------------------------------------------------------------------------
__global__ void cvt_f32_bf16_kernel(const float* __restrict__ src,
                                    bf16* __restrict__ dst, size_t n) {
  size_t i = (size_t)blockIdx.x * blockDim.x + threadIdx.x;
  size_t stride = (size_t)gridDim.x * blockDim.x;
  for (; i < n; i += stride) dst[i] = (bf16)src[i];
}

// pad-convert: src [M,K] f32 -> dst [M,Kp] bf16, zero fill K..Kp
__global__ void cvt_pad_kernel(const float* __restrict__ src, bf16* __restrict__ dst,
                               int K, int Kp, size_t n /* M*Kp */) {
  size_t i = (size_t)blockIdx.x * blockDim.x + threadIdx.x;
  size_t stride = (size_t)gridDim.x * blockDim.x;
  for (; i < n; i += stride) {
    size_t m = i / Kp;
    int k = (int)(i - m * Kp);
    dst[i] = (k < K) ? (bf16)src[m * K + k] : (bf16)0.f;
  }
}

// batched transpose-convert: src [Bt,R,C] f32 -> dst [Bt,C,Rp] bf16, zero pad
__global__ void cvt_tr_kernel(const float* __restrict__ src, bf16* __restrict__ dst,
                              int R, int C, int Rp, size_t n /* Bt*C*Rp */) {
  size_t i = (size_t)blockIdx.x * blockDim.x + threadIdx.x;
  size_t stride = (size_t)gridDim.x * blockDim.x;
  for (; i < n; i += stride) {
    size_t bt = i / ((size_t)C * Rp);
    size_t rem = i - bt * (size_t)C * Rp;
    int c = (int)(rem / Rp), r = (int)(rem - (size_t)c * Rp);
    dst[i] = (r < R) ? (bf16)src[(bt * R + r) * C + c] : (bf16)0.f;
  }
}

// ---------------------------------------------------------------------------
// Branch-free WMMA GEMM: C[M,N] = A[M,Kp] * B[N,Kp]-major + bias.
// Invariants: M % 128 == 0, N % 64 == 0, K % 32 == 0, buffers padded.
// 128x64 block tile, 8 waves, each wave 32x32 (4 WMMAs / k-step).
// Double-buffered LDS, staged by async global->LDS b128 copies.
// ---------------------------------------------------------------------------
__global__ __launch_bounds__(256) void gemm_bf16_kernel(
    const bf16* __restrict__ A, int lda,
    const bf16* __restrict__ Bm, int ldb,
    const float* __restrict__ bias,
    void* __restrict__ C, int ldc, int c_bf16, int K) {
  __shared__ __align__(16) bf16 sA[2][128 * 32];
  __shared__ __align__(16) bf16 sBt[2][64 * 40];  // [n][k], row stride 40

  int m0 = blockIdx.y * 128;
  int n0 = blockIdx.x * 64;
  int tid = threadIdx.x;
  int w = tid >> 5, lane = tid & 31;
  int wr = (w & 3) * 32;
  int wc = (w >> 2) * 32;

  // per-thread staging coordinates
  int ar0 = tid >> 2, ach0 = (tid & 3) << 3;          // A chunk 0
  int ar1 = (tid + 256) >> 2, ach1 = ach0;            // A chunk 1
  int bn = tid >> 2, bch = (tid & 3) << 3;            // B chunk

  v8f acc00 = {}, acc01 = {}, acc10 = {}, acc11 = {};
  int Ksteps = K >> 5;

  // prologue: stage tile 0 into buffer 0
  async_copy_b128(&A[(size_t)(m0 + ar0) * lda + ach0], &sA[0][ar0 * 32 + ach0]);
  async_copy_b128(&A[(size_t)(m0 + ar1) * lda + ach1], &sA[0][ar1 * 32 + ach1]);
  async_copy_b128(&Bm[(size_t)(n0 + bn) * ldb + bch], &sBt[0][bn * 40 + bch]);
  wait_async0();
  __syncthreads();

  for (int ks = 0; ks < Ksteps; ++ks) {
    int cur = ks & 1, nxt = cur ^ 1;
    if (ks + 1 < Ksteps) {
      int k1 = (ks + 1) << 5;
      async_copy_b128(&A[(size_t)(m0 + ar0) * lda + k1 + ach0], &sA[nxt][ar0 * 32 + ach0]);
      async_copy_b128(&A[(size_t)(m0 + ar1) * lda + k1 + ach1], &sA[nxt][ar1 * 32 + ach1]);
      async_copy_b128(&Bm[(size_t)(n0 + bn) * ldb + k1 + bch], &sBt[nxt][bn * 40 + bch]);
      if (ks + 2 < Ksteps && tid < 128)
        __builtin_prefetch(&A[(size_t)(m0 + tid) * lda + k1 + 32], 0, 1);
    }
    v16bf a0 = load_frag_a(&sA[cur][wr * 32], 32);
    v16bf a1 = load_frag_a(&sA[cur][(wr + 16) * 32], 32);
    v16bf b0 = load_frag_bk(&sBt[cur][wc * 40], 40);
    v16bf b1 = load_frag_bk(&sBt[cur][(wc + 16) * 40], 40);
    acc00 = wmma_bf16(a0, b0, acc00);
    acc01 = wmma_bf16(a0, b1, acc01);
    acc10 = wmma_bf16(a1, b0, acc10);
    acc11 = wmma_bf16(a1, b1, acc11);
    if (ks + 1 < Ksteps) wait_async0();
    __syncthreads();
  }

  int n = lane & 15, rb = (lane >> 4) << 3;
  v8f* accs[4] = {&acc00, &acc01, &acc10, &acc11};
#pragma unroll
  for (int f = 0; f < 4; ++f) {
    int mi2 = f >> 1, ni2 = f & 1;
    int gn = n0 + wc + ni2 * 16 + n;
    float bv = bias ? bias[gn] : 0.f;
    CDu u; u.v = *accs[f];
#pragma unroll
    for (int r = 0; r < 8; ++r) {
      int gm = m0 + wr + mi2 * 16 + rb + r;
      float v = u.e[r] + bv;
      if (c_bf16) ((bf16*)C)[(size_t)gm * ldc + gn] = (bf16)v;
      else ((float*)C)[(size_t)gm * ldc + gn] = v;
    }
  }
}

// ---------------------------------------------------------------------------
// Flash attention with negated probs: ctx = colsum(V) - softmax(QK^T/8) V
// ---------------------------------------------------------------------------
__global__ __launch_bounds__(256) void attn_neg_kernel(
    const bf16* __restrict__ Q, const bf16* __restrict__ K,
    const bf16* __restrict__ V, bf16* __restrict__ CTX) {
  const int S = 512, D = 128;
  __shared__ __align__(16) bf16 sQ[64 * 64];
  __shared__ __align__(16) bf16 sK[64 * 64];
  __shared__ __align__(16) bf16 sVt[64 * 72];  // [d][key]
  __shared__ __align__(16) float sS[64 * 64];
  __shared__ __align__(16) bf16 sP[64 * 64];
  __shared__ float rowm[64], rowl[64], rowc[64], vsum[64];
  __shared__ float red[64 * 4];

  int q0 = blockIdx.x * 64;
  int b = blockIdx.y >> 1, h = blockIdx.y & 1;
  const size_t base = (size_t)b * S * D + (size_t)h * 64;
  int tid = threadIdx.x;
  int w = tid >> 5, lane = tid & 31;
  int mi = w & 3;
  int ni0 = (w >> 2) * 2, ni1 = ni0 + 1;
  int row = tid & 63, part = tid >> 6;

  for (int i = tid; i < 64 * 64; i += 256) {
    int r = i >> 6, c = i & 63;
    sQ[i] = Q[base + (size_t)(q0 + r) * D + c];
  }
  if (tid < 64) { rowm[tid] = -1e30f; rowl[tid] = 0.f; vsum[tid] = 0.f; }
  v8f o0 = {}, o1 = {};
  __syncthreads();

  for (int kb = 0; kb < 8; ++kb) {
    int k0 = kb * 64;
    for (int i = tid; i < 64 * 64; i += 256) {
      int r = i >> 6, c = i & 63;
      sK[i] = K[base + (size_t)(k0 + r) * D + c];
      sVt[c * 72 + r] = V[base + (size_t)(k0 + r) * D + c];
    }
    __syncthreads();

    v8f s0 = {}, s1 = {};
#pragma unroll
    for (int kk = 0; kk < 2; ++kk) {
      v16bf aq  = load_frag_a(&sQ[(mi * 16) * 64 + kk * 32], 64);
      v16bf bk0 = load_frag_bk(&sK[(ni0 * 16) * 64 + kk * 32], 64);
      v16bf bk1 = load_frag_bk(&sK[(ni1 * 16) * 64 + kk * 32], 64);
      s0 = wmma_bf16(aq, bk0, s0);
      s1 = wmma_bf16(aq, bk1, s1);
    }
    {
      CDu u0, u1; u0.v = s0; u1.v = s1;
      int n = lane & 15, rb = (lane >> 4) << 3;
#pragma unroll
      for (int r = 0; r < 8; ++r) {
        sS[(mi * 16 + rb + r) * 64 + ni0 * 16 + n] = u0.e[r] * 0.125f;
        sS[(mi * 16 + rb + r) * 64 + ni1 * 16 + n] = u1.e[r] * 0.125f;
      }
    }
    __syncthreads();

    float pmax = -1e30f;
    for (int c = part * 16; c < part * 16 + 16; ++c)
      pmax = fmaxf(pmax, sS[row * 64 + c]);
    red[row * 4 + part] = pmax;
    __syncthreads();
    if (part == 0) {
      float mold = rowm[row];
      float mm = fmaxf(fmaxf(red[row * 4], red[row * 4 + 1]),
                       fmaxf(red[row * 4 + 2], red[row * 4 + 3]));
      float mnew = fmaxf(mold, mm);
      rowm[row] = mnew;
      rowc[row] = __expf(mold - mnew);
    }
    __syncthreads();
    float mnew = rowm[row];
    float psum = 0.f;
    for (int c = part * 16; c < part * 16 + 16; ++c) {
      float p = __expf(sS[row * 64 + c] - mnew);
      sP[row * 64 + c] = (bf16)p;
      psum += p;
    }
    red[row * 4 + part] = psum;
    __syncthreads();
    if (part == 0)
      rowl[row] = rowl[row] * rowc[row] +
                  (red[row * 4] + red[row * 4 + 1] + red[row * 4 + 2] + red[row * 4 + 3]);
    __syncthreads();
    float vs = 0.f;
    for (int r = part * 16; r < part * 16 + 16; ++r) vs += (float)sVt[row * 72 + r];
    red[row * 4 + part] = vs;
    __syncthreads();
    if (part == 0)
      vsum[row] += red[row * 4] + red[row * 4 + 1] + red[row * 4 + 2] + red[row * 4 + 3];
    __syncthreads();

    {
      int rb = (lane >> 4) << 3;
      CDu u0, u1; u0.v = o0; u1.v = o1;
#pragma unroll
      for (int r = 0; r < 8; ++r) {
        float c = rowc[mi * 16 + rb + r];
        u0.e[r] *= c; u1.e[r] *= c;
      }
      o0 = u0.v; o1 = u1.v;
    }
#pragma unroll
    for (int kk = 0; kk < 2; ++kk) {
      v16bf ap  = load_frag_a(&sP[(mi * 16) * 64 + kk * 32], 64);
      v16bf bv0 = load_frag_bk(&sVt[(ni0 * 16) * 72 + kk * 32], 72);
      v16bf bv1 = load_frag_bk(&sVt[(ni1 * 16) * 72 + kk * 32], 72);
      o0 = wmma_bf16(ap, bv0, o0);
      o1 = wmma_bf16(ap, bv1, o1);
    }
    __syncthreads();
  }

  {
    int n = lane & 15, rb = (lane >> 4) << 3;
    CDu u0, u1; u0.v = o0; u1.v = o1;
#pragma unroll
    for (int r = 0; r < 8; ++r) {
      int rr = mi * 16 + rb + r;
      float inv_l = 1.f / rowl[rr];
      int gs = q0 + rr;
      int d0 = ni0 * 16 + n, d1 = ni1 * 16 + n;
      CTX[base + (size_t)gs * D + d0] = (bf16)(vsum[d0] - u0.e[r] * inv_l);
      CTX[base + (size_t)gs * D + d1] = (bf16)(vsum[d1] - u1.e[r] * inv_l);
    }
  }
}

// ---------------------------------------------------------------------------
__global__ __launch_bounds__(128) void layernorm_kernel(
    const float* __restrict__ X, const float* __restrict__ g,
    const float* __restrict__ bta, bf16* __restrict__ Y) {
  __shared__ float red[128];
  int row = blockIdx.x, tid = threadIdx.x;
  float v = X[(size_t)row * 128 + tid];
  red[tid] = v; __syncthreads();
  for (int s = 64; s > 0; s >>= 1) { if (tid < s) red[tid] += red[tid + s]; __syncthreads(); }
  float mean = red[0] * (1.f / 128.f);
  __syncthreads();
  float d = v - mean;
  red[tid] = d * d; __syncthreads();
  for (int s = 64; s > 0; s >>= 1) { if (tid < s) red[tid] += red[tid + s]; __syncthreads(); }
  float var = red[0] * (1.f / 128.f);
  Y[(size_t)row * 128 + tid] = (bf16)(d * rsqrtf(var + 1e-5f) * g[tid] + bta[tid]);
}

__global__ void avg2_bf16_kernel(const bf16* __restrict__ a, const bf16* __restrict__ b,
                                 bf16* __restrict__ o, size_t n) {
  size_t i = (size_t)blockIdx.x * blockDim.x + threadIdx.x;
  size_t stride = (size_t)gridDim.x * blockDim.x;
  for (; i < n; i += stride)
    o[i] = (bf16)(0.5f * ((float)a[i] + (float)b[i]));
}

// ---------------------------------------------------------------------------
// Persistent single-direction GRU. 512 threads (16 waves), one block per
// (modality, direction). Per step: gh = h@Whh^T, gi = x_t@Wih^T via WMMA,
// then fused gates. Dynamic LDS 224KB (<= 320KB/WGP).
// ---------------------------------------------------------------------------
__global__ __launch_bounds__(512) void gru_dir_kernel(
    const bf16* __restrict__ X, const bf16* __restrict__ Wih,
    const bf16* __restrict__ Whh,
    const float* __restrict__ bih, const float* __restrict__ bhh,
    bf16* __restrict__ HS, int dir) {
  const int S = 512, D = 128, B = 64, G = 384;
  extern __shared__ __align__(16) char smem[];
  bf16* sH = (bf16*)smem;
  bf16* sX = sH + B * D;
  float* sGh = (float*)(smem + 2 * B * D * sizeof(bf16));
  float* sGi = sGh + B * G;

  int tid = threadIdx.x;
  int w = tid >> 5, lane = tid & 31;
  for (int i = tid; i < B * D; i += 512) sH[i] = (bf16)0.f;
  __syncthreads();

  for (int t = 0; t < S; ++t) {
    int ti = dir ? (S - 1 - t) : t;
    for (int i = tid; i < B * D; i += 512) {
      int bb = i >> 7, j = i & 127;
      sX[i] = X[((size_t)bb * S + ti) * D + j];
    }
    __syncthreads();

#pragma unroll
    for (int j6 = 0; j6 < 6; ++j6) {
      int tt = w * 6 + j6;
      int tmi = tt & 3, tni = tt >> 2;
      v8f ah = {}, ai = {};
#pragma unroll
      for (int kk = 0; kk < 4; ++kk) {
        v16bf fa_h  = load_frag_a(&sH[(tmi * 16) * D + kk * 32], D);
        v16bf fa_x  = load_frag_a(&sX[(tmi * 16) * D + kk * 32], D);
        v16bf fb_hh = load_frag_bk(&Whh[(size_t)(tni * 16) * D + kk * 32], D);
        v16bf fb_ih = load_frag_bk(&Wih[(size_t)(tni * 16) * D + kk * 32], D);
        ah = wmma_bf16(fa_h, fb_hh, ah);
        ai = wmma_bf16(fa_x, fb_ih, ai);
      }
      int n = lane & 15, rb = (lane >> 4) << 3;
      CDu uh, ui; uh.v = ah; ui.v = ai;
#pragma unroll
      for (int r = 0; r < 8; ++r) {
        sGh[(tmi * 16 + rb + r) * G + tni * 16 + n] = uh.e[r];
        sGi[(tmi * 16 + rb + r) * G + tni * 16 + n] = ui.e[r];
      }
    }
    __syncthreads();

    for (int i = tid; i < B * D; i += 512) {
      int bb = i >> 7, j = i & 127;
      float xr = sGi[bb * G + j]       + bih[j];
      float xz = sGi[bb * G + 128 + j] + bih[128 + j];
      float xn = sGi[bb * G + 256 + j] + bih[256 + j];
      float hr = sGh[bb * G + j]       + bhh[j];
      float hz = sGh[bb * G + 128 + j] + bhh[128 + j];
      float hn = sGh[bb * G + 256 + j] + bhh[256 + j];
      float rg = 1.f / (1.f + __expf(-(xr + hr)));
      float zg = 1.f / (1.f + __expf(-(xz + hz)));
      float ng = tanhf(xn + rg * hn);
      float hnew = (1.f - zg) * ng + zg * (float)sH[i];
      sH[i] = (bf16)hnew;
      HS[((size_t)bb * S + ti) * D + j] = (bf16)hnew;
    }
    __syncthreads();
  }
}

__global__ __launch_bounds__(128) void seqmean_kernel(const bf16* __restrict__ hs,
                                                      float* __restrict__ out) {
  int b = blockIdx.x, buf = blockIdx.y, j = threadIdx.x;
  const size_t BSD = (size_t)64 * 512 * 128;
  const bf16* p = hs + buf * BSD + (size_t)b * 512 * 128 + j;
  float s = 0.f;
  for (int t = 0; t < 512; ++t) s += (float)p[(size_t)t * 128];
  out[((size_t)buf * 64 + b) * 128 + j] = s * (1.f / 512.f);
}

__global__ void combine_kernel(const float* __restrict__ means, bf16* __restrict__ out) {
  int i = blockIdx.x * blockDim.x + threadIdx.x;
  if (i >= 64 * 384) return;
  int b = i / 384, c = i % 384;
  int seg = c >> 7, j = c & 127;
  float x = 0.5f * (means[((size_t)seg * 64 + b) * 128 + j] +
                    means[((size_t)(seg + 3) * 64 + b) * 128 + j]);
  out[i] = (bf16)x;
}

__global__ void bnrelu6_kernel(const float* __restrict__ x, const float* __restrict__ g,
                               const float* __restrict__ b, bf16* __restrict__ y, int n, int C) {
  int i = blockIdx.x * blockDim.x + threadIdx.x;
  if (i >= n) return;
  int c = i % C;
  float v = x[i] * rsqrtf(1.f + 1e-5f) * g[c] + b[c];
  v = fminf(fmaxf(v, 0.f), 6.f);
  y[i] = (bf16)v;
}

// extract [64,8] from padded [128,64] GEMM output and add final bias
__global__ void head_out_kernel(const float* __restrict__ h2,
                                const float* __restrict__ fb2,
                                float* __restrict__ out) {
  int i = blockIdx.x * blockDim.x + threadIdx.x;
  if (i >= 64 * 8) return;
  int b = i >> 3, o = i & 7;
  out[i] = h2[b * 64 + o] + fb2[o];
}

// ---------------------------------------------------------------------------
extern "C" void kernel_launch(void* const* d_in, const int* in_sizes, int n_in,
                              void* d_out, int out_size, void* d_ws, size_t ws_size,
                              hipStream_t stream) {
  (void)in_sizes; (void)n_in; (void)out_size; (void)ws_size;
  const size_t BS = (size_t)64 * 512;
  const size_t BSD = BS * 128;
  const int KT = 320, KV = 64, KA = 96;  // padded feature K dims

  char* ws = (char*)d_ws;
  size_t off = 0;
  auto alloc = [&](size_t bytes) -> char* {
    char* p = ws + off;
    off = (off + bytes + 255) & ~(size_t)255;
    return p;
  };

  bf16* wFc1t = (bf16*)alloc((size_t)128 * KT * 2);
  bf16* wFc2t = (bf16*)alloc((size_t)128 * KV * 2);
  bf16* wFc3t = (bf16*)alloc((size_t)128 * KA * 2);
  bf16* wQt = (bf16*)alloc((size_t)6 * 128 * 128 * 2);
  bf16* wKt = (bf16*)alloc((size_t)6 * 128 * 128 * 2);
  bf16* wVt = (bf16*)alloc((size_t)6 * 128 * 128 * 2);
  bf16* wDt = (bf16*)alloc((size_t)6 * 128 * 128 * 2);
  bf16* wIh = (bf16*)alloc((size_t)6 * 384 * 128 * 2);
  bf16* wHh = (bf16*)alloc((size_t)6 * 384 * 128 * 2);
  bf16* wF1t = (bf16*)alloc((size_t)256 * 384 * 2);
  bf16* wF2t = (bf16*)alloc((size_t)64 * 256 * 2);  // rows 8..63 unused
  bf16* fT = (bf16*)alloc(BS * KT * 2);
  bf16* fV = (bf16*)alloc(BS * KV * 2);
  bf16* fA = (bf16*)alloc(BS * KA * 2);
  bf16* tB = (bf16*)alloc(BSD * 2);
  bf16* vB = (bf16*)alloc(BSD * 2);
  bf16* aB = (bf16*)alloc(BSD * 2);
  bf16* qB = (bf16*)alloc(BSD * 2);
  bf16* kB = (bf16*)alloc(BSD * 2);
  bf16* vpB = (bf16*)alloc(BSD * 2);
  bf16* ctxB = (bf16*)alloc(BSD * 2);
  bf16* lnout = (bf16*)alloc(6 * BSD * 2);
  bf16* xT = (bf16*)alloc(BSD * 2);
  bf16* xV = (bf16*)alloc(BSD * 2);
  bf16* xA = (bf16*)alloc(BSD * 2);
  bf16* hsAll = (bf16*)alloc(6 * BSD * 2);
  float* projTmp = (float*)alloc(BSD * 4);
  float* means = (float*)alloc(6 * 64 * 128 * 4);
  bf16* mcomb = (bf16*)alloc((size_t)128 * 384 * 2);
  float* h1t = (float*)alloc((size_t)128 * 256 * 4);
  bf16* h1b = (bf16*)alloc((size_t)128 * 256 * 2);
  float* h2t = (float*)alloc((size_t)128 * 64 * 4);

  auto nblk = [](size_t n) {
    size_t b = (n + 2047) / 2048;
    return (int)(b > 4096 ? 4096 : (b < 1 ? 1 : b));
  };
  auto cvt = [&](const void* s, bf16* dst, size_t n) {
    cvt_f32_bf16_kernel<<<nblk(n), 256, 0, stream>>>((const float*)s, dst, n);
  };
  auto cvt_tr = [&](const void* s, bf16* dst, int Bt, int R, int C, int Rp) {
    size_t n = (size_t)Bt * C * Rp;
    cvt_tr_kernel<<<nblk(n), 256, 0, stream>>>((const float*)s, dst, R, C, Rp, n);
  };
  auto cvt_pad = [&](const void* s, bf16* dst, size_t M, int K, int Kp) {
    size_t n = M * Kp;
    cvt_pad_kernel<<<nblk(n), 256, 0, stream>>>((const float*)s, dst, K, Kp, n);
  };
  auto gemm = [&](const bf16* A, int lda, const bf16* Bm, int ldb,
                  const float* bias, void* C, int ldc, int c_bf16,
                  int M, int N, int K) {
    dim3 grid(N / 64, M / 128);
    gemm_bf16_kernel<<<grid, 256, 0, stream>>>(A, lda, Bm, ldb, bias, C, ldc,
                                               c_bf16, K);
  };

  cvt_tr(d_in[3], wFc1t, 1, 300, 128, KT);
  cvt_tr(d_in[5], wFc2t, 1, 35, 128, KV);
  cvt_tr(d_in[7], wFc3t, 1, 74, 128, KA);
  cvt_tr(d_in[9], wQt, 6, 128, 128, 128);
  cvt_tr(d_in[11], wKt, 6, 128, 128, 128);
  cvt_tr(d_in[13], wVt, 6, 128, 128, 128);
  cvt_tr(d_in[15], wDt, 6, 128, 128, 128);
  cvt(d_in[19], wIh, (size_t)6 * 384 * 128);
  cvt(d_in[20], wHh, (size_t)6 * 384 * 128);
  cvt_tr(d_in[23], wF1t, 1, 384, 256, 384);
  cvt_tr(d_in[27], wF2t, 1, 256, 8, 256);
  cvt_pad(d_in[0], fT, BS, 300, KT);
  cvt_pad(d_in[1], fV, BS, 35, KV);
  cvt_pad(d_in[2], fA, BS, 74, KA);

  gemm(fT, KT, wFc1t, KT, (const float*)d_in[4], tB, 128, 1, (int)BS, 128, KT);
  gemm(fV, KV, wFc2t, KV, (const float*)d_in[6], vB, 128, 1, (int)BS, 128, KV);
  gemm(fA, KA, wFc3t, KA, (const float*)d_in[8], aB, 128, 1, (int)BS, 128, KA);

  const bf16* uq[6] = {tB, aB, tB, vB, vB, aB};
  const bf16* ukv[6] = {aB, tB, vB, tB, aB, vB};
  const float* bq = (const float*)d_in[10];
  const float* bk = (const float*)d_in[12];
  const float* bv = (const float*)d_in[14];
  const float* bd = (const float*)d_in[16];
  const float* lng = (const float*)d_in[17];
  const float* lnb = (const float*)d_in[18];

  for (int u = 0; u < 6; ++u) {
    gemm(uq[u], 128, wQt + (size_t)u * 128 * 128, 128, bq + u * 128, qB, 128, 1, (int)BS, 128, 128);
    gemm(ukv[u], 128, wKt + (size_t)u * 128 * 128, 128, bk + u * 128, kB, 128, 1, (int)BS, 128, 128);
    gemm(ukv[u], 128, wVt + (size_t)u * 128 * 128, 128, bv + u * 128, vpB, 128, 1, (int)BS, 128, 128);
    attn_neg_kernel<<<dim3(8, 128), 256, 0, stream>>>(qB, kB, vpB, ctxB);
    gemm(ctxB, 128, wDt + (size_t)u * 128 * 128, 128, bd + u * 128, projTmp, 128, 0, (int)BS, 128, 128);
    layernorm_kernel<<<(int)BS, 128, 0, stream>>>(projTmp, lng + u * 128, lnb + u * 128,
                                                  lnout + (size_t)u * BSD);
  }

  avg2_bf16_kernel<<<4096, 256, 0, stream>>>(lnout + 1 * BSD, lnout + 3 * BSD, xT, BSD);
  avg2_bf16_kernel<<<4096, 256, 0, stream>>>(lnout + 2 * BSD, lnout + 5 * BSD, xV, BSD);
  avg2_bf16_kernel<<<4096, 256, 0, stream>>>(lnout + 0 * BSD, lnout + 4 * BSD, xA, BSD);

  {
    const bf16* xs[3] = {xT, xV, xA};
    const float* gbih = (const float*)d_in[21];
    const float* gbhh = (const float*)d_in[22];
    size_t smem = 2 * 64 * 128 * sizeof(bf16) + 2 * (size_t)64 * 384 * sizeof(float);
    for (int m = 0; m < 3; ++m)
      for (int d = 0; d < 2; ++d) {
        int idx = m * 2 + d;
        gru_dir_kernel<<<1, 512, smem, stream>>>(
            xs[m], wIh + (size_t)idx * 384 * 128, wHh + (size_t)idx * 384 * 128,
            gbih + idx * 384, gbhh + idx * 384, hsAll + (size_t)idx * BSD, d);
      }
  }

  seqmean_kernel<<<dim3(64, 6), 128, 0, stream>>>(hsAll, means);
  combine_kernel<<<(64 * 384 + 255) / 256, 256, 0, stream>>>(means, mcomb);

  gemm(mcomb, 384, wF1t, 384, (const float*)d_in[24], h1t, 256, 0, 128, 256, 384);
  bnrelu6_kernel<<<(64 * 256 + 255) / 256, 256, 0, stream>>>(
      h1t, (const float*)d_in[25], (const float*)d_in[26], h1b, 64 * 256, 256);
  gemm(h1b, 256, wF2t, 256, nullptr, h2t, 64, 0, 128, 64, 256);
  head_out_kernel<<<2, 256, 0, stream>>>(h2t, (const float*)d_in[28], (float*)d_out);
}
